// Performer__40862318854891
// MI455X (gfx1250) — matrix operand order repaired and checked by reading, compiled, not run on hardware
//
#include <hip/hip_runtime.h>
#include <hip/hip_bf16.h>
#include <cstdint>

// ---------------------------------------------------------------------------
// Performer forward for MI455X (gfx1250, wave32).
// Big GEMMs: v_wmma_f32_16x16x32_bf16, 128x128 block tile, 8 WMMA / K-step,
// double-buffered LDS tiles staged with global_load_async_to_lds_b128
// (ASYNCcnt) so DMA overlaps WMMA — CDNA5-specific data path.
// Scan: chunked linear attention, 266x64 f32 state in LDS (68KB dynamic,
// legal on CDNA5's 320KB/WGP).
// ---------------------------------------------------------------------------

typedef unsigned short u16;
typedef __attribute__((ext_vector_type(16))) __bf16 v16bf;
typedef __attribute__((ext_vector_type(8)))  float  v8f;

#define B_    2
#define N_    2048
#define DIM_  1024
#define H_    16
#define DH_   64
#define MP_   266          // int(DH * ln(DH))
#define FF_   4096
#define DEPTH_ 6
#define ROWS_ (B_*N_)      // 4096
#define CH_   128          // scan chunk length
#define NC_   (N_/CH_)     // 16 chunks
#define SE_   (MP_*DH_)    // 17024 state elements
#define JT_   ((SE_+255)/256)  // 67 accum slots / thread

#define DN_F    0.35355339059327373f   // DH^-0.25
#define DNSQ_F  0.125f                 // DN^2
#define RATIO_F 0.06131393394849658f   // MP^-0.5

__device__ __forceinline__ u16 f2bf(float f) {
  unsigned u = __float_as_uint(f);
  u += 0x7FFFu + ((u >> 16) & 1u);       // round-to-nearest-even
  return (u16)(u >> 16);
}

__device__ __forceinline__ float gelu_f(float x) {
  const float k0 = 0.7978845608028654f, k1 = 0.044715f;
  return 0.5f * x * (1.0f + tanhf(k0 * (x + k1 * x * x * x)));
}

union Frag { uint4 u4[2]; v16bf bf; };

// CDNA5 async copy global -> LDS (16B per lane), tracked with ASYNCcnt.
// Flat LDS addresses carry the LDS offset in their low 32 bits (ISA 10.2).
__device__ __forceinline__ void async_ld_b128(void* lds_dst, const void* gsrc) {
  unsigned loff = (unsigned)(unsigned long long)lds_dst;
  asm volatile("global_load_async_to_lds_b128 %0, %1, off"
               :: "v"(loff), "v"(gsrc) : "memory");
}
__device__ __forceinline__ void wait_async0() {
  asm volatile("s_wait_asynccnt 0" ::: "memory");
}

// ---------------------------------------------------------------------------
// f32 -> bf16 elementwise convert (activation-side GEMM operands)
// ---------------------------------------------------------------------------
__global__ __launch_bounds__(256) void cvt_bf16_kernel(
    const float* __restrict__ src, u16* __restrict__ dst, size_t n) {
  size_t i = (size_t)blockIdx.x * 256 + threadIdx.x;
  if (i < n) dst[i] = f2bf(src[i]);
}

// ---------------------------------------------------------------------------
// f32 [K,N] -> bf16 [N,K] transpose-convert (weight-side GEMM operands),
// LDS-tiled 32x32 so both global sides stay coalesced.
// grid = (N/32, K/32), block = 256.
// ---------------------------------------------------------------------------
__global__ __launch_bounds__(256) void cvt_t_bf16_kernel(
    const float* __restrict__ in, u16* __restrict__ out, int K, int Nn) {
  __shared__ float tile[32][33];
  int tx = threadIdx.x & 31, ty = threadIdx.x >> 5;   // 8 rows per pass
  int nb = blockIdx.x * 32, kb = blockIdx.y * 32;
  for (int r = ty; r < 32; r += 8)
    tile[r][tx] = in[(size_t)(kb + r) * Nn + nb + tx];
  __syncthreads();
  for (int r = ty; r < 32; r += 8)
    out[(size_t)(nb + r) * K + kb + tx] = f2bf(tile[tx][r]);
}

// ---------------------------------------------------------------------------
// LayerNorm over dim (divisible by 256). Writes bf16 and/or f32 output.
// ---------------------------------------------------------------------------
__global__ __launch_bounds__(256) void ln_kernel(
    const float* __restrict__ x, const float* __restrict__ g,
    const float* __restrict__ bb, u16* __restrict__ obf,
    float* __restrict__ of32, int dim) {
  __shared__ float r1[256], r2[256];
  const float* row = x + (size_t)blockIdx.x * dim;
  float s = 0.f, s2 = 0.f;
  for (int j = threadIdx.x; j < dim; j += 256) {
    float v = row[j]; s += v; s2 += v * v;
  }
  r1[threadIdx.x] = s; r2[threadIdx.x] = s2; __syncthreads();
  for (int st = 128; st > 0; st >>= 1) {
    if (threadIdx.x < st) { r1[threadIdx.x] += r1[threadIdx.x+st]; r2[threadIdx.x] += r2[threadIdx.x+st]; }
    __syncthreads();
  }
  float mu  = r1[0] / (float)dim;
  float var = r2[0] / (float)dim - mu * mu;
  float inv = rsqrtf(var + 1e-5f);
  for (int j = threadIdx.x; j < dim; j += 256) {
    float v = (row[j] - mu) * inv * g[j] + bb[j];
    size_t o = (size_t)blockIdx.x * dim + j;
    if (obf)  obf[o]  = f2bf(v);
    if (of32) of32[o] = v;
  }
}

// ---------------------------------------------------------------------------
// WMMA bf16 GEMM: C[M,N] = A[M,K] @ B[K,N] (+bias)(gelu)(+residual)
// A: [M,K] bf16 row-major.  Bt: [N,K] bf16 (pre-transposed weights).
// Block 256 threads = 8 waves; block tile 128x128, K-step 32.
// Wave grid 4(M) x 2(N); wave tile 32x64 = 2x4 accumulators, 8 WMMA / step.
// Double-buffered: next tile's global_load_async_to_lds_b128 issued before
// computing the current tile; s_wait_asynccnt + barrier afterwards both
// publishes the next tile and protects the just-read buffer from overwrite.
// M,N,K must divide tiles (true here: 4096 x {1024,4096} x {1024,4096}).
// Fragment layouts per CDNA5 ISA 7.12.2 (wave32); rows padded to 40 halves
// to spread the 16-lane b128 fragment gathers across LDS banks.
// ---------------------------------------------------------------------------
__global__ __launch_bounds__(256) void gemm_bf16_kernel(
    const u16* __restrict__ A, const u16* __restrict__ Bt,
    float* __restrict__ C, const float* __restrict__ bias,
    const float* __restrict__ residual, int M, int Nn, int K, int fuse_gelu) {
  __shared__ __align__(16) u16 sA[2][128][40];
  __shared__ __align__(16) u16 sB[2][128][40];
  const int tid  = threadIdx.x;
  const int wave = tid >> 5, lane = tid & 31;
  const int wm = wave & 3, wn = wave >> 2;
  const int row0 = blockIdx.y * 128, col0 = blockIdx.x * 128;
  const int l16 = lane & 15, hi = lane >> 4;

  // Stage A (128x32) and Bt (128x32): 512 16B chunks each, async to LDS.
  auto stage = [&](int buf, int k0) {
    #pragma unroll
    for (int ch = tid; ch < 512; ch += 256) {
      int r = ch >> 2, cc = (ch & 3) * 8;
      async_ld_b128(&sA[buf][r][cc], A  + (size_t)(row0 + r) * K + k0 + cc);
      async_ld_b128(&sB[buf][r][cc], Bt + (size_t)(col0 + r) * K + k0 + cc);
    }
  };

  v8f acc[2][4] = {};

  stage(0, 0);
  wait_async0();
  __syncthreads();

  int buf = 0;
  for (int k0 = 0; k0 < K; k0 += 32) {
    if (k0 + 32 < K) stage(buf ^ 1, k0 + 32);   // overlap DMA with WMMA

    // A frag (16x32 MxK): lane<16 holds K 0-7 & 16-23; lane>=16: 8-15 & 24-31
    Frag a[2];
    #pragma unroll
    for (int i = 0; i < 2; ++i) {
      const u16* pr = &sA[buf][wm * 32 + i * 16 + l16][0];
      a[i].u4[0] = *(const uint4*)(pr + hi * 8);
      a[i].u4[1] = *(const uint4*)(pr + hi * 8 + 16);
    }
    // B frag (32x16 KxN): lane<16 holds K 0-7 & 16-23 of column n; lane>=16: 8-15 & 24-31
    Frag b[4];
    #pragma unroll
    for (int j = 0; j < 4; ++j) {
      const u16* pr = &sB[buf][wn * 64 + j * 16 + l16][0];
      b[j].u4[0] = *(const uint4*)(pr + hi * 8);
      b[j].u4[1] = *(const uint4*)(pr + hi * 8 + 16);
    }
    #pragma unroll
    for (int i = 0; i < 2; ++i)
      #pragma unroll
      for (int j = 0; j < 4; ++j)
        acc[i][j] = __builtin_amdgcn_wmma_f32_16x16x32_bf16(
            false, a[i].bf, false, b[j].bf, (short)0, acc[i][j], false, false);

    wait_async0();     // own next-tile DMA complete
    __syncthreads();   // all waves: next tile visible, current tile fully read
    buf ^= 1;
  }

  // C/D layout: lane l<16 -> N=l, M = hi*8 + vgpr index
  #pragma unroll
  for (int i = 0; i < 2; ++i) {
    #pragma unroll
    for (int j = 0; j < 4; ++j) {
      int nn = col0 + wn * 64 + j * 16 + l16;
      float bv = bias ? bias[nn] : 0.f;
      #pragma unroll
      for (int r = 0; r < 8; ++r) {
        int m = row0 + wm * 32 + i * 16 + hi * 8 + r;
        float v = acc[i][j][r] + bv;
        if (fuse_gelu) v = gelu_f(v);
        if (residual)  v += residual[(size_t)m * Nn + nn];
        C[(size_t)m * Nn + nn] = v;
      }
    }
  }
}

// ---------------------------------------------------------------------------
// Rotary embedding, in place on q and k ([B*N, DIM] head-sliced layout).
// ---------------------------------------------------------------------------
__global__ __launch_bounds__(256) void rotary_kernel(
    float* __restrict__ q, float* __restrict__ k, const float* __restrict__ pe) {
  size_t i = (size_t)blockIdx.x * 256 + threadIdx.x;
  const size_t total = (size_t)ROWS_ * DIM_ / 2;
  if (i >= total) return;
  int half = (int)(i % (DH_ / 2));       // 0..31
  size_t bn = i / (DIM_ / 2);            // b*N + n
  int n = (int)(bn % N_);
  float s  = pe[(size_t)n * DH_ + half];
  float cc = pe[(size_t)n * DH_ + DH_ / 2 + half];
  size_t base = i * 2;
  float q0 = q[base], q1 = q[base + 1];
  q[base]     = q0 * cc - q1 * s;
  q[base + 1] = q1 * cc + q0 * s;
  float k0 = k[base], k1 = k[base + 1];
  k[base]     = k0 * cc - k1 * s;
  k[base + 1] = k1 * cc + k0 * s;
}

// ---------------------------------------------------------------------------
// dd = (data @ proj^T) * dn, diag = 0.5*dn^2*sum(d^2), per-row max.
// One block per (b,h,n); data is [B*N, DIM] head-sliced, proj is [MP, DH].
// ---------------------------------------------------------------------------
__global__ __launch_bounds__(256) void dd_kernel(
    const float* __restrict__ data, const float* __restrict__ proj,
    float* __restrict__ dd, float* __restrict__ diag, float* __restrict__ rowmax) {
  __shared__ float sd[DH_];
  __shared__ float red[256];
  int idx = blockIdx.x;                       // bh*N + n
  int n = idx % N_, h = (idx / N_) % H_, b = idx / (N_ * H_);
  const float* row = data + ((size_t)(b * N_ + n)) * DIM_ + h * DH_;
  if (threadIdx.x < DH_) sd[threadIdx.x] = row[threadIdx.x];
  __syncthreads();
  float dp = (threadIdx.x < DH_) ? sd[threadIdx.x] * sd[threadIdx.x] : 0.f;
  red[threadIdx.x] = dp; __syncthreads();
  for (int st = 128; st > 0; st >>= 1) {
    if (threadIdx.x < st) red[threadIdx.x] += red[threadIdx.x + st];
    __syncthreads();
  }
  float dg = red[0] * 0.5f * DNSQ_F;
  __syncthreads();
  float mx = -3.4e38f;
  float* out = dd + (size_t)idx * MP_;
  for (int m = threadIdx.x; m < MP_; m += 256) {
    const float* pm = proj + (size_t)m * DH_;
    float acc = 0.f;
    #pragma unroll 8
    for (int j = 0; j < DH_; ++j) acc += sd[j] * pm[j];
    acc *= DN_F;
    out[m] = acc;
    mx = fmaxf(mx, acc);
  }
  red[threadIdx.x] = mx; __syncthreads();
  for (int st = 128; st > 0; st >>= 1) {
    if (threadIdx.x < st) red[threadIdx.x] = fmaxf(red[threadIdx.x], red[threadIdx.x + st]);
    __syncthreads();
  }
  if (threadIdx.x == 0) { rowmax[idx] = red[0]; diag[idx] = dg; }
}

// per-(b,h) max over n of rowmax (key branch)
__global__ __launch_bounds__(256) void bhmax_kernel(
    const float* __restrict__ rowmax, float* __restrict__ bhmax) {
  __shared__ float red[256];
  float m = -3.4e38f;
  for (int n = threadIdx.x; n < N_; n += 256)
    m = fmaxf(m, rowmax[(size_t)blockIdx.x * N_ + n]);
  red[threadIdx.x] = m; __syncthreads();
  for (int st = 128; st > 0; st >>= 1) {
    if (threadIdx.x < st) red[threadIdx.x] = fmaxf(red[threadIdx.x], red[threadIdx.x + st]);
    __syncthreads();
  }
  if (threadIdx.x == 0) bhmax[blockIdx.x] = red[0];
}

// dd -> ratio*(exp(dd - diag - mx) + 1e-4), in place
__global__ __launch_bounds__(256) void expapply_kernel(
    float* __restrict__ dd, const float* __restrict__ diag,
    const float* __restrict__ mx, int per_row) {
  size_t i = (size_t)blockIdx.x * 256 + threadIdx.x;
  const size_t total = (size_t)B_ * H_ * N_ * MP_;
  if (i >= total) return;
  size_t rowi = i / MP_;
  size_t bh = rowi / N_;
  float m = per_row ? mx[rowi] : mx[bh];
  dd[i] = RATIO_F * (expf(dd[i] - diag[rowi] - m) + 1e-4f);
}

// ---------------------------------------------------------------------------
// Scan pass 1: per-chunk KV outer-product sums (register accumulators).
// grid = B*H*NC.  cS[bh,c,m,e], cZ[bh,c,m]
// ---------------------------------------------------------------------------
__global__ __launch_bounds__(256) void chunk_sum_kernel(
    const float* __restrict__ pk, const float* __restrict__ vb,
    float* __restrict__ cS, float* __restrict__ cZ) {
  __shared__ float k_s[MP_];
  __shared__ float v_s[DH_];
  int c = blockIdx.x % NC_, bh = blockIdx.x / NC_;
  int b = bh / H_, h = bh % H_;
  float acc[JT_];
  #pragma unroll
  for (int j = 0; j < JT_; ++j) acc[j] = 0.f;
  float z0 = 0.f, z1 = 0.f;
  for (int tt = 0; tt < CH_; ++tt) {
    int t = c * CH_ + tt;
    for (int m = threadIdx.x; m < MP_; m += 256)
      k_s[m] = pk[((size_t)bh * N_ + t) * MP_ + m];
    if (threadIdx.x < DH_)
      v_s[threadIdx.x] = vb[((size_t)(b * N_ + t)) * DIM_ + h * DH_ + threadIdx.x];
    __syncthreads();
    #pragma unroll
    for (int j = 0; j < JT_; ++j) {
      int idx = threadIdx.x + j * 256;
      if (idx < SE_) acc[j] += k_s[idx >> 6] * v_s[idx & 63];
    }
    if (threadIdx.x < MP_)       z0 += k_s[threadIdx.x];
    if (threadIdx.x < MP_ - 256) z1 += k_s[threadIdx.x + 256];
    __syncthreads();
  }
  float* oS = cS + ((size_t)bh * NC_ + c) * SE_;
  #pragma unroll
  for (int j = 0; j < JT_; ++j) {
    int idx = threadIdx.x + j * 256;
    if (idx < SE_) oS[idx] = acc[j];
  }
  float* oZ = cZ + ((size_t)bh * NC_ + c) * MP_;
  if (threadIdx.x < MP_)       oZ[threadIdx.x] = z0;
  if (threadIdx.x < MP_ - 256) oZ[threadIdx.x + 256] = z1;
}

// Pass 2: inclusive prefix over chunks, per (b,h). Element i is always owned
// by the same thread across c, so no barriers needed.
__global__ __launch_bounds__(256) void prefix_kernel(
    float* __restrict__ cS, float* __restrict__ cZ) {
  size_t base = (size_t)blockIdx.x * NC_;
  for (int c = 1; c < NC_; ++c) {
    float* cur = cS + (base + c) * SE_;
    const float* prev = cS + (base + c - 1) * SE_;
    for (int i = threadIdx.x; i < SE_; i += 256) cur[i] += prev[i];
    float* curz = cZ + (base + c) * MP_;
    const float* prevz = cZ + (base + c - 1) * MP_;
    for (int i = threadIdx.x; i < MP_; i += 256) curz[i] += prevz[i];
  }
}

// Pass 3: per-chunk causal scan seeded with exclusive prefix; S in dynamic LDS.
// grid = B*H*NC, dynamic shared = SE_*4 bytes (68KB, fine on 320KB WGP LDS).
__global__ __launch_bounds__(256) void scan_kernel(
    const float* __restrict__ pq, const float* __restrict__ pk,
    const float* __restrict__ vb, const float* __restrict__ cS,
    const float* __restrict__ cZ, float* __restrict__ ob) {
  extern __shared__ float S[];                 // [MP_*DH_]
  __shared__ float z_s[MP_];
  __shared__ float q_s[MP_], k_s[MP_], v_s[DH_];
  __shared__ float red[256], outred[256];
  int c = blockIdx.x % NC_, bh = blockIdx.x / NC_;
  int b = bh / H_, h = bh % H_;

  if (c == 0) {
    for (int i = threadIdx.x; i < SE_; i += 256) S[i] = 0.f;
    for (int i = threadIdx.x; i < MP_; i += 256) z_s[i] = 0.f;
  } else {
    const float* ps = cS + ((size_t)bh * NC_ + c - 1) * SE_;
    for (int i = threadIdx.x; i < SE_; i += 256) S[i] = ps[i];
    const float* pz = cZ + ((size_t)bh * NC_ + c - 1) * MP_;
    for (int i = threadIdx.x; i < MP_; i += 256) z_s[i] = pz[i];
  }
  __syncthreads();

  const int e = threadIdx.x & 63, mg = threadIdx.x >> 6;
  for (int tt = 0; tt < CH_; ++tt) {
    int t = c * CH_ + tt;
    for (int m = threadIdx.x; m < MP_; m += 256) {
      q_s[m] = pq[((size_t)bh * N_ + t) * MP_ + m];
      k_s[m] = pk[((size_t)bh * N_ + t) * MP_ + m];
    }
    if (threadIdx.x < DH_)
      v_s[threadIdx.x] = vb[((size_t)(b * N_ + t)) * DIM_ + h * DH_ + threadIdx.x];
    __syncthreads();
    // S += k (x) v ; z += k
    #pragma unroll
    for (int j = 0; j < JT_; ++j) {
      int idx = threadIdx.x + j * 256;
      if (idx < SE_) S[idx] += k_s[idx >> 6] * v_s[idx & 63];
    }
    for (int m = threadIdx.x; m < MP_; m += 256) z_s[m] += k_s[m];
    __syncthreads();
    // denom = q . (z + eps); out[e] = (q . S[:,e]) / denom
    float dp = 0.f;
    for (int m = threadIdx.x; m < MP_; m += 256) dp += q_s[m] * (z_s[m] + 1e-6f);
    red[threadIdx.x] = dp;
    float op = 0.f;
    for (int m = mg; m < MP_; m += 4) op += q_s[m] * S[m * DH_ + e];
    outred[threadIdx.x] = op;
    __syncthreads();
    for (int st = 128; st > 0; st >>= 1) {
      if (threadIdx.x < st) red[threadIdx.x] += red[threadIdx.x + st];
      __syncthreads();
    }
    float denom = red[0];
    if (threadIdx.x < DH_) {
      float o = outred[threadIdx.x] + outred[threadIdx.x + 64] +
                outred[threadIdx.x + 128] + outred[threadIdx.x + 192];
      ob[((size_t)(b * N_ + t)) * DIM_ + h * DH_ + threadIdx.x] = o / denom;
    }
    __syncthreads();
  }
}

// ---------------------------------------------------------------------------
// Host orchestration
// ---------------------------------------------------------------------------
static inline void run_cvt(const float* s, u16* d, size_t n, hipStream_t st) {
  cvt_bf16_kernel<<<(unsigned)((n + 255) / 256), 256, 0, st>>>(s, d, n);
}
static inline void run_cvt_t(const float* s, u16* d, int K, int Nn, hipStream_t st) {
  dim3 g(Nn / 32, K / 32);
  cvt_t_bf16_kernel<<<g, 256, 0, st>>>(s, d, K, Nn);
}
static inline void run_gemm(const u16* A, const u16* Bt, float* C,
                            const float* bias, const float* res,
                            int M, int Nn, int K, int gelu, hipStream_t st) {
  dim3 g(Nn / 128, M / 128);
  gemm_bf16_kernel<<<g, 256, 0, st>>>(A, Bt, C, bias, res, M, Nn, K, gelu);
}

extern "C" void kernel_launch(void* const* d_in, const int* in_sizes, int n_in,
                              void* d_out, int out_size, void* d_ws, size_t ws_size,
                              hipStream_t stream) {
  (void)in_sizes; (void)n_in; (void)out_size; (void)ws_size;
  const float* x_in = (const float*)d_in[0];
  const float* pe   = (const float*)d_in[1];
  const float* Wq   = (const float*)d_in[2];
  const float* Wk   = (const float*)d_in[3];
  const float* Wv   = (const float*)d_in[4];
  const float* Wo   = (const float*)d_in[5];
  const float* proj = (const float*)d_in[6];
  const float* ln1g = (const float*)d_in[7];
  const float* ln1b = (const float*)d_in[8];
  const float* ln2g = (const float*)d_in[9];
  const float* ln2b = (const float*)d_in[10];
  const float* W1   = (const float*)d_in[11];
  const float* b1   = (const float*)d_in[12];
  const float* W2   = (const float*)d_in[13];
  const float* b2   = (const float*)d_in[14];
  const float* lnfg = (const float*)d_in[15];
  const float* lnfb = (const float*)d_in[16];

  // bump allocator over d_ws (~370 MB total)
  char* p = (char*)d_ws;
  auto alloc = [&](size_t bytes) -> void* {
    void* r = (void*)p;
    p += (bytes + 255) & ~((size_t)255);
    return r;
  };
  const size_t RD = (size_t)ROWS_ * DIM_;
  float* xcur  = (float*)alloc(RD * 4);
  u16*   ybf   = (u16*)  alloc(RD * 2);
  u16*   wbuf  = (u16*)  alloc((size_t)DIM_ * FF_ * 2);     // largest weight (4M)
  float* qb    = (float*)alloc(RD * 4);
  float* kb    = (float*)alloc(RD * 4);
  float* vb    = (float*)alloc(RD * 4);
  float* pqb   = (float*)alloc((size_t)B_ * H_ * N_ * MP_ * 4);
  float* pkb   = (float*)alloc((size_t)B_ * H_ * N_ * MP_ * 4);
  float* diagq = (float*)alloc((size_t)B_ * H_ * N_ * 4);
  float* diagk = (float*)alloc((size_t)B_ * H_ * N_ * 4);
  float* rmq   = (float*)alloc((size_t)B_ * H_ * N_ * 4);
  float* rmk   = (float*)alloc((size_t)B_ * H_ * N_ * 4);
  float* bhm   = (float*)alloc((size_t)B_ * H_ * 4);
  float* cS    = (float*)alloc((size_t)B_ * H_ * NC_ * SE_ * 4);
  float* cZ    = (float*)alloc((size_t)B_ * H_ * NC_ * MP_ * 4);
  float* ob    = (float*)alloc(RD * 4);
  u16*   obf   = (u16*)  alloc(RD * 2);
  float* hdn   = (float*)alloc((size_t)ROWS_ * FF_ * 4);
  u16*   hdnbf = (u16*)  alloc((size_t)ROWS_ * FF_ * 2);

  hipMemcpyAsync(xcur, x_in, RD * sizeof(float), hipMemcpyDeviceToDevice, stream);

  const size_t W_QKV = (size_t)DIM_ * DIM_;
  const size_t W_FF  = (size_t)DIM_ * FF_;
  const int nBHN = B_ * H_ * N_;
  const unsigned expg = (unsigned)(((size_t)nBHN * MP_ + 255) / 256);

  for (int l = 0; l < DEPTH_; ++l) {
    // y = LN1(x) -> bf16
    ln_kernel<<<ROWS_, 256, 0, stream>>>(xcur, ln1g + (size_t)l * DIM_,
                                         ln1b + (size_t)l * DIM_, ybf, nullptr, DIM_);
    // q, k, v projections (WMMA bf16, weights pre-transposed to [N,K])
    run_cvt_t(Wq + l * W_QKV, wbuf, DIM_, DIM_, stream);
    run_gemm(ybf, wbuf, qb, nullptr, nullptr, ROWS_, DIM_, DIM_, 0, stream);
    run_cvt_t(Wk + l * W_QKV, wbuf, DIM_, DIM_, stream);
    run_gemm(ybf, wbuf, kb, nullptr, nullptr, ROWS_, DIM_, DIM_, 0, stream);
    run_cvt_t(Wv + l * W_QKV, wbuf, DIM_, DIM_, stream);
    run_gemm(ybf, wbuf, vb, nullptr, nullptr, ROWS_, DIM_, DIM_, 0, stream);
    // rotary (in place, f32)
    rotary_kernel<<<(unsigned)((RD / 2 + 255) / 256), 256, 0, stream>>>(qb, kb, pe);
    // softmax kernel features (f32 for exp precision)
    const float* pj = proj + (size_t)l * MP_ * DH_;
    dd_kernel<<<nBHN, 256, 0, stream>>>(qb, pj, pqb, diagq, rmq);
    dd_kernel<<<nBHN, 256, 0, stream>>>(kb, pj, pkb, diagk, rmk);
    bhmax_kernel<<<B_ * H_, 256, 0, stream>>>(rmk, bhm);
    expapply_kernel<<<expg, 256, 0, stream>>>(pqb, diagq, rmq, 1);
    expapply_kernel<<<expg, 256, 0, stream>>>(pkb, diagk, bhm, 0);
    // chunked causal linear attention
    chunk_sum_kernel<<<B_ * H_ * NC_, 256, 0, stream>>>(pkb, vb, cS, cZ);
    prefix_kernel<<<B_ * H_, 256, 0, stream>>>(cS, cZ);
    scan_kernel<<<B_ * H_ * NC_, 256, SE_ * sizeof(float), stream>>>(
        pqb, pkb, vb, cS, cZ, ob);
    // x += o @ Wo   (residual fused in epilogue)
    run_cvt(ob, obf, RD, stream);
    run_cvt_t(Wo + l * W_QKV, wbuf, DIM_, DIM_, stream);
    run_gemm(obf, wbuf, xcur, nullptr, xcur, ROWS_, DIM_, DIM_, 0, stream);
    // FFN: x += GELU(LN2(x) @ W1 + b1) @ W2 + b2
    ln_kernel<<<ROWS_, 256, 0, stream>>>(xcur, ln2g + (size_t)l * DIM_,
                                         ln2b + (size_t)l * DIM_, ybf, nullptr, DIM_);
    run_cvt_t(W1 + (size_t)l * W_FF, wbuf, DIM_, FF_, stream);
    run_gemm(ybf, wbuf, hdn, b1 + (size_t)l * FF_, nullptr, ROWS_, FF_, DIM_, 1, stream);
    run_cvt(hdn, hdnbf, (size_t)ROWS_ * FF_, stream);
    run_cvt_t(W2 + (size_t)l * W_FF, wbuf, FF_, DIM_, stream);
    run_gemm(hdnbf, wbuf, xcur, b2 + (size_t)l * DIM_, xcur, ROWS_, DIM_, FF_, 0, stream);
  }
  // final layernorm -> f32 output
  ln_kernel<<<ROWS_, 256, 0, stream>>>(xcur, lnfg, lnfb, nullptr, (float*)d_out, DIM_);
}
